// LSTM_27462020891002
// MI455X (gfx1250) — matrix operand order repaired and checked by reading, not compiled, not run on hardware
//
#include <hip/hip_runtime.h>
#include <stdint.h>

#define HID 1024
#define SEQ 512
#define BAT 32
#define G4  4096
#define LDSW 272   // padded LDS row stride (floats): 272 % 64 == 16 -> conflict-free

typedef __bf16 bf16_t;
typedef __attribute__((ext_vector_type(16))) __bf16 v16bf;
typedef __attribute__((ext_vector_type(8)))  float  v8f;

struct alignas(16) U128 { uint32_t x[4]; };
union V16U { v16bf v; U128 q[2]; };

// ---------------- WMMA fragment loaders (CDNA5 16x16x32 bf16 layouts) ----------------
// A matrix 16x32 (MxK), row-major source with leading dim ld.
// lane L: M = L&15, half = L>>4.
//   elements 0..7  -> K = k0 + 8*half + j        (contiguous 16B)
//   elements 8..15 -> K = k0 + 16 + 8*half + j   (contiguous 16B)
static __device__ inline v16bf load_a_bf16(const bf16_t* __restrict__ src, int ld,
                                           int row0, int k0, int lane) {
  const int m = lane & 15, hf = lane >> 4;
  const bf16_t* p = src + (size_t)(row0 + m) * ld + (k0 + hf * 8);
  V16U u;
  u.q[0] = *(const U128*)(p);
  u.q[1] = *(const U128*)(p + 16);
  return u.v;
}

// A fragment gathered from an fp32 source (converted in-register to bf16).
static __device__ inline v16bf load_a_f32(const float* __restrict__ src, int ld,
                                          int row0, int k0, int lane) {
  const int m = lane & 15, hf = lane >> 4;
  const float* p = src + (size_t)(row0 + m) * ld + (k0 + hf * 8);
  v16bf a;
#pragma unroll
  for (int j = 0; j < 8; ++j) {
    a[j]     = (bf16_t)p[j];
    a[j + 8] = (bf16_t)p[16 + j];
  }
  return a;
}

// B matrix 32x16 (KxN) where B[k][n] = W[n][k], W row-major (N x K), leading dim ld.
// lane L: N = L&15, half = L>>4; elements j=0..7 -> K = k0 + 8*half + j, j=8..15 -> K = k0 + 16 + 8*half + (j-8)
static __device__ inline v16bf load_b_bf16(const bf16_t* __restrict__ W, int ld,
                                           int n0, int k0, int lane) {
  const int n = lane & 15, hf = lane >> 4;
  const bf16_t* p = W + (size_t)(n0 + n) * ld + (k0 + hf * 8);
  V16U u;
  u.q[0] = *(const U128*)(p);
  u.q[1] = *(const U128*)(p + 16);
  return u.v;
}

static __device__ inline v8f wmma_bf16(v16bf a, v16bf b, v8f c) {
  return __builtin_amdgcn_wmma_f32_16x16x32_bf16(false, a, false, b, (short)0, c,
                                                 false, false);
}

static __device__ inline float sigmoidf_(float x) { return 1.0f / (1.0f + __expf(-x)); }

// ---------------- conversion / init kernels ----------------
__global__ void k_f2bf(const float* __restrict__ src, bf16_t* __restrict__ dst, int n) {
  int i = blockIdx.x * 256 + threadIdx.x;
  if (i < n) dst[i] = (bf16_t)src[i];
}

// x: (B,T,H) fp32  ->  xbf: (T,B,H) bf16 (time-major rows match pre/hs row order)
__global__ void k_x_to_bf(const float* __restrict__ x, bf16_t* __restrict__ xbf) {
  int i = blockIdx.x * 256 + threadIdx.x;                // 0 .. B*T*H-1
  int j  = i & (HID - 1);
  int bt = i >> 10;                                      // b*T + t
  int t  = bt & (SEQ - 1);
  int b  = bt >> 9;
  xbf[((size_t)t * BAT + b) * HID + j] = (bf16_t)x[i];
}

__global__ void k_init(const float* __restrict__ b_ih_f, const float* __restrict__ b_hh_f,
                       const float* __restrict__ b_ih_b, const float* __restrict__ b_hh_b,
                       float* __restrict__ bcomb, float* __restrict__ cstate,
                       bf16_t* __restrict__ h0, bf16_t* __restrict__ h1) {
  int i = blockIdx.x * 256 + threadIdx.x;
  if (i < G4) {
    bcomb[i]      = b_ih_f[i] + b_hh_f[i];
    bcomb[G4 + i] = b_ih_b[i] + b_hh_b[i];
  }
  if (i < 2 * BAT * HID) {
    cstate[i] = 0.0f;
    h0[i]     = (bf16_t)0.0f;
    h1[i]     = (bf16_t)0.0f;
  }
}

// ---------------- time-parallel input-projection GEMM ----------------
// pre[d][t*B+b][n] (bf16) = sum_k xbf[t*B+b][k] * W_ih_d[n][k]
// M = T*B = 16384, N = 4096, K = 1024, per direction. Wave tile: 16x64.
__global__ void k_pre_gemm(const bf16_t* __restrict__ xbf,
                           const bf16_t* __restrict__ Wih_f,
                           const bf16_t* __restrict__ Wih_b,
                           bf16_t* __restrict__ pre_f, bf16_t* __restrict__ pre_b) {
  int wave = (int)((blockIdx.x * 256u + threadIdx.x) >> 5);
  int lane = threadIdx.x & 31;
  int dir  = wave >> 16;            // 65536 waves per direction
  int rem  = wave & 65535;
  int mt   = rem >> 6;              // 0..1023
  int ng   = rem & 63;              // 0..63
  const bf16_t* W = dir ? Wih_b : Wih_f;
  bf16_t* pre     = dir ? pre_b : pre_f;
  const int m0 = mt * 16, n0 = ng * 64;

  v8f acc[4] = {};
  for (int k0 = 0; k0 < HID; k0 += 32) {
    v16bf a = load_a_bf16(xbf, HID, m0, k0, lane);
#pragma unroll
    for (int i = 0; i < 4; ++i) {
      v16bf b = load_b_bf16(W, HID, n0 + i * 16, k0, lane);
      acc[i] = wmma_bf16(a, b, acc[i]);
    }
  }
  const int hf = lane >> 4, nl = lane & 15;
#pragma unroll
  for (int i = 0; i < 4; ++i)
#pragma unroll
    for (int v = 0; v < 8; ++v) {
      int row = m0 + v + 8 * hf;
      int n   = n0 + i * 16 + nl;
      pre[(size_t)row * G4 + n] = (bf16_t)acc[i][v];
    }
}

// ---------------- fused recurrent step (GEMM -> LDS -> LSTM pointwise) ----------------
// 32 blocks: dir = blk>>4, hidden slice j0 = (blk&15)*64.
// Each block: gates(b, {g*H + j0..j0+63}) = h @ W_hh^T slice  (M=32,N=256,K=1024),
// gate tile staged in LDS, then LSTM cell update for (b, j0..j0+63) in-block.
// h state is double-buffered across steps (hin -> hout) to avoid cross-block races.
__global__ void k_step_fused(const bf16_t* __restrict__ hin,
                             const bf16_t* __restrict__ Whh_f,
                             const bf16_t* __restrict__ Whh_b,
                             const bf16_t* __restrict__ pre_f,
                             const bf16_t* __restrict__ pre_b,
                             const float* __restrict__ bcomb,
                             float* __restrict__ cstate,
                             bf16_t* __restrict__ hout,
                             float* __restrict__ hs_f, float* __restrict__ hs_b,
                             int t) {
  __shared__ float lgates[BAT][LDSW];   // [batch][4 gates x 64 cols], padded stride

  const int dir = blockIdx.x >> 4;
  const int j0  = (blockIdx.x & 15) * 64;
  const bf16_t* W = dir ? Whh_b : Whh_f;
  const bf16_t* h = hin + (size_t)dir * BAT * HID;

  const int wid  = threadIdx.x >> 5;    // 0..7
  const int lane = threadIdx.x & 31;
  const int mt = wid & 1;               // m-tile: batch rows 0-15 / 16-31
  const int ng = wid >> 1;              // gate index 0..3 (i,f,g,o)
  const int m0 = mt * 16;
  const int n0 = ng * HID + j0;         // global gate-column base for this wave

  v8f acc[4] = {};
  for (int k0 = 0; k0 < HID; k0 += 32) {
    v16bf a = load_a_bf16(h, HID, m0, k0, lane);
#pragma unroll
    for (int i = 0; i < 4; ++i) {
      v16bf b = load_b_bf16(W, HID, n0 + i * 16, k0, lane);
      acc[i] = wmma_bf16(a, b, acc[i]);
    }
  }
  const int hf = lane >> 4, nl = lane & 15;
#pragma unroll
  for (int i = 0; i < 4; ++i)
#pragma unroll
    for (int v = 0; v < 8; ++v)
      lgates[m0 + v + 8 * hf][ng * 64 + i * 16 + nl] = acc[i][v];
  __syncthreads();

  // LSTM cell update: 32 batch x 64 hidden units = 2048 items, 8 per thread
  const int tt = dir ? (SEQ - 1 - t) : t;     // backward scans reversed time
  const bf16_t* pre = dir ? pre_b : pre_f;
  const float* bb = bcomb + (size_t)dir * G4;
  float* hs = dir ? hs_b : hs_f;
#pragma unroll
  for (int it = 0; it < 8; ++it) {
    int item = threadIdx.x + it * 256;
    int b  = item >> 6;
    int jj = item & 63;
    int j  = j0 + jj;
    size_t prow = ((size_t)tt * BAT + b) * G4;
    float gi = lgates[b][jj]       + (float)pre[prow + j]        + bb[j];         // i
    float gf = lgates[b][64 + jj]  + (float)pre[prow + j + 1024] + bb[j + 1024];  // f
    float gz = lgates[b][128 + jj] + (float)pre[prow + j + 2048] + bb[j + 2048];  // g
    float go = lgates[b][192 + jj] + (float)pre[prow + j + 3072] + bb[j + 3072];  // o
    size_t cidx = (size_t)dir * BAT * HID + (size_t)b * HID + j;
    float c = sigmoidf_(gf) * cstate[cidx] + sigmoidf_(gi) * tanhf(gz);
    cstate[cidx] = c;
    float hh = sigmoidf_(go) * tanhf(c);
    hout[cidx] = (bf16_t)hh;
    hs[((size_t)tt * BAT + b) * HID + j] = hh;
  }
}

// ---------------- fused gating GEMM + output ----------------
// sig = sigmoid([hs_f|hs_b] @ Wg^T + bg); out(B,T,H) = sig*hs_f + (1-sig)*hs_b
// M = T*B = 16384, N = 1024, K = 2048. Wave tile 16x64, epilogue fused.
__global__ void k_gate_fused(const float* __restrict__ hs_f,
                             const float* __restrict__ hs_b,
                             const bf16_t* __restrict__ Wg,
                             const float* __restrict__ bg,
                             float* __restrict__ out) {
  int wave = (int)((blockIdx.x * 256u + threadIdx.x) >> 5);
  int lane = threadIdx.x & 31;
  int mt = wave >> 4;               // 0..1023
  int ng = wave & 15;               // 0..15
  const int m0 = mt * 16, n0 = ng * 64;

  v8f acc[4] = {};
  for (int k0 = 0; k0 < 2 * HID; k0 += 32) {
    const float* src = (k0 < HID) ? hs_f : hs_b;
    int kk = (k0 < HID) ? k0 : (k0 - HID);
    v16bf a = load_a_f32(src, HID, m0, kk, lane);
#pragma unroll
    for (int i = 0; i < 4; ++i) {
      v16bf b = load_b_bf16(Wg, 2 * HID, n0 + i * 16, k0, lane);
      acc[i] = wmma_bf16(a, b, acc[i]);
    }
  }
  const int hf = lane >> 4, nl = lane & 15;
#pragma unroll
  for (int i = 0; i < 4; ++i)
#pragma unroll
    for (int v = 0; v < 8; ++v) {
      int row = m0 + v + 8 * hf;        // t*B + b
      int t = row >> 5;
      int b = row & 31;
      int j = n0 + i * 16 + nl;
      float s  = sigmoidf_(acc[i][v] + bg[j]);
      float vf = hs_f[(size_t)row * HID + j];
      float vb = hs_b[(size_t)row * HID + j];
      out[((size_t)b * SEQ + t) * HID + j] = s * vf + (1.0f - s) * vb;
    }
}

// ---------------- host launcher ----------------
extern "C" void kernel_launch(void* const* d_in, const int* in_sizes, int n_in,
                              void* d_out, int out_size, void* d_ws, size_t ws_size,
                              hipStream_t stream) {
  (void)in_sizes; (void)n_in; (void)out_size; (void)ws_size;
  const float* x      = (const float*)d_in[0];
  const float* W_ih_f = (const float*)d_in[1];
  const float* W_hh_f = (const float*)d_in[2];
  const float* b_ih_f = (const float*)d_in[3];
  const float* b_hh_f = (const float*)d_in[4];
  const float* W_ih_b = (const float*)d_in[5];
  const float* W_hh_b = (const float*)d_in[6];
  const float* b_ih_b = (const float*)d_in[7];
  const float* b_hh_b = (const float*)d_in[8];
  const float* Wg     = (const float*)d_in[9];
  const float* bg     = (const float*)d_in[10];
  float* out = (float*)d_out;

  char* ws = (char*)d_ws;
  size_t off = 0;
  auto take = [&](size_t bytes) -> char* {
    char* p = ws + off;
    off += (bytes + 255) & ~(size_t)255;
    return p;
  };
  bf16_t* Wihf_bf = (bf16_t*)take((size_t)G4 * HID * 2);
  bf16_t* Wihb_bf = (bf16_t*)take((size_t)G4 * HID * 2);
  bf16_t* Whhf_bf = (bf16_t*)take((size_t)G4 * HID * 2);
  bf16_t* Whhb_bf = (bf16_t*)take((size_t)G4 * HID * 2);
  bf16_t* Wg_bf   = (bf16_t*)take((size_t)HID * 2 * HID * 2);
  bf16_t* xbf     = (bf16_t*)take((size_t)SEQ * BAT * HID * 2);
  bf16_t* pre_f   = (bf16_t*)take((size_t)SEQ * BAT * G4 * 2);
  bf16_t* pre_b   = (bf16_t*)take((size_t)SEQ * BAT * G4 * 2);
  float*  hs_f    = (float*)take((size_t)SEQ * BAT * HID * 4);
  float*  hs_b    = (float*)take((size_t)SEQ * BAT * HID * 4);
  float*  bcomb   = (float*)take((size_t)2 * G4 * 4);
  float*  cstate  = (float*)take((size_t)2 * BAT * HID * 4);
  bf16_t* h0      = (bf16_t*)take((size_t)2 * BAT * HID * 2);
  bf16_t* h1      = (bf16_t*)take((size_t)2 * BAT * HID * 2);

  // fp32 -> bf16 weight conversion
  k_f2bf<<<16384, 256, 0, stream>>>(W_ih_f, Wihf_bf, G4 * HID);
  k_f2bf<<<16384, 256, 0, stream>>>(W_ih_b, Wihb_bf, G4 * HID);
  k_f2bf<<<16384, 256, 0, stream>>>(W_hh_f, Whhf_bf, G4 * HID);
  k_f2bf<<<16384, 256, 0, stream>>>(W_hh_b, Whhb_bf, G4 * HID);
  k_f2bf<<<8192,  256, 0, stream>>>(Wg, Wg_bf, HID * 2 * HID);
  k_init<<<256, 256, 0, stream>>>(b_ih_f, b_hh_f, b_ih_b, b_hh_b, bcomb, cstate, h0, h1);
  k_x_to_bf<<<65536, 256, 0, stream>>>(x, xbf);

  // hoisted, time-parallel input-projection GEMMs (both directions)
  k_pre_gemm<<<16384, 256, 0, stream>>>(xbf, Wihf_bf, Wihb_bf, pre_f, pre_b);

  // sequential recurrence: one fused kernel per step, h double-buffered
  for (int t = 0; t < SEQ; ++t) {
    const bf16_t* hin = (t & 1) ? h1 : h0;
    bf16_t*      hout = (t & 1) ? h0 : h1;
    k_step_fused<<<32, 256, 0, stream>>>(hin, Whhf_bf, Whhb_bf, pre_f, pre_b,
                                         bcomb, cstate, hout, hs_f, hs_b, t);
  }

  // fused gating GEMM + convex-combination output
  k_gate_fused<<<2048, 256, 0, stream>>>(hs_f, hs_b, Wg_bf, bg, out);
}
